// TransformerDecoderLayer_51419348468345
// MI455X (gfx1250) — matrix operand order repaired and checked by reading, not compiled, not run on hardware
//
#include <hip/hip_runtime.h>
#include <hip/hip_bf16.h>
#include <math.h>

// ---------------------------------------------------------------------------
// Types
// ---------------------------------------------------------------------------
typedef __attribute__((ext_vector_type(16))) __bf16 v16bf;
typedef __attribute__((ext_vector_type(8)))  float  v8f;
typedef long long i64;

#define B_  4
#define L_  1024
#define D_  512
#define H_  8
#define F_  2048

union FragU { v16bf v; float4 f4[2]; };

// WGP-scope prefetch (scope field = 0 -> pull into all cache levels, stop on
// WGP$ hit). clang's __builtin_prefetch only emits SE/SYS scopes, which per
// the ISA prefetch table do NOT fill the WGP cache -- useless for L2-resident
// tiles. Addresses passed here are always inside allocated tensors.
__device__ __forceinline__ void prefetch_wgp(const void* p) {
    asm volatile("global_prefetch_b8 %0, off" :: "v"(p));
}

// A fragment (16x32 bf16, M x K): lane holds row m = lane&15.
// lanes 0-15: K = {0..7, 16..23}; lanes 16-31: K = {8..15, 24..31}
__device__ __forceinline__ v16bf load_a_frag(const __bf16* rowp, int laneHi) {
    FragU f;
    const __bf16* p = rowp + laneHi * 8;
    f.f4[0] = *(const float4*)(p);
    f.f4[1] = *(const float4*)(p + 16);
    return f.v;
}
// B fragment (32x16 bf16, K x N), NT layout: lane holds col n = lane&15 (a row
// of the second source matrix). lanes 0-15: K = {0..7, 16..23}; lanes 16-31: K = {8..15, 24..31}
__device__ __forceinline__ v16bf load_b_frag(const __bf16* rowp, int laneHi) {
    FragU f;
    const __bf16* p = rowp + laneHi * 8;
    f.f4[0] = *(const float4*)(p);
    f.f4[1] = *(const float4*)(p + 16);
    return f.v;
}

// ---------------------------------------------------------------------------
// Epilogues
// ---------------------------------------------------------------------------
struct EpiBf16 {            // C[z][m][n] -> bf16
    __bf16* C; i64 cB; int ldc;
    __device__ __forceinline__ void store(int z, int m, int n, float v) const {
        C[(i64)z * cB + (i64)m * ldc + n] = (__bf16)v;
    }
};
struct EpiBf16T {           // transposed store: C[z][n][m] -> bf16 (for V^T)
    __bf16* C; i64 cB; int ldc;
    __device__ __forceinline__ void store(int z, int m, int n, float v) const {
        C[(i64)z * cB + (i64)n * ldc + m] = (__bf16)v;
    }
};
struct EpiScaleF32 {        // logits: C[z][m][n] = v*scale (f32, into d_out)
    float* C; i64 cB; int ldc; float scale;
    __device__ __forceinline__ void store(int z, int m, int n, float v) const {
        C[(i64)z * cB + (i64)m * ldc + n] = v * scale;
    }
};
struct EpiComb {            // ctx -> comb with torch concat/view permutation
    __bf16* C;              // comb [B][L][H*D]
    __device__ __forceinline__ void store(int z, int m, int n, float v) const {
        int h = z & 7, b = z >> 3;
        int kk = (((n >> 6) << 3) + h) * 64 + (n & 63);
        C[((i64)b * L_ + m) * (i64)(H_ * D_) + kk] = (__bf16)v;
    }
};
struct EpiResF32 {          // t = tokens + sa  (f32 into d_out)
    float* C; const float* R; i64 cB; int ldc;
    __device__ __forceinline__ void store(int z, int m, int n, float v) const {
        i64 idx = (i64)z * cB + (i64)m * ldc + n;
        C[idx] = R[idx] + v;
    }
};
struct EpiBiasGelu {        // h1 = gelu_exact(v + b1[n]) -> bf16
    __bf16* C; const float* bias; i64 cB; int ldc;
    __device__ __forceinline__ void store(int z, int m, int n, float v) const {
        float x = v + bias[n];
        x = 0.5f * x * (1.0f + erff(x * 0.70710678118654752f));
        C[(i64)z * cB + (i64)m * ldc + n] = (__bf16)x;
    }
};
struct EpiFFNOut {          // t += v + b2[n]; also emit bf16 copy of t
    float* T; __bf16* Tbf; const float* bias; i64 cB; int ldc;
    __device__ __forceinline__ void store(int z, int m, int n, float v) const {
        i64 idx = (i64)z * cB + (i64)m * ldc + n;
        float x = T[idx] + v + bias[n];
        T[idx] = x;
        Tbf[idx] = (__bf16)x;
    }
};

// ---------------------------------------------------------------------------
// Core NT GEMM: C[m,n] = sum_k A[m,k]*B[n,k], A/B bf16, wave-level WMMA.
// Block = 256 threads = 8 waves; block tile 128(M) x 128(N);
// wave tile 64(M) x 32(N) = 4x2 WMMA tiles. Dims must be multiples of 128/32.
// WGP-scope prefetch runs 2 K-steps ahead; its guard is scalar-uniform so
// EXEC remains all-ones across the WMMAs (required by the ISA).
// ---------------------------------------------------------------------------
template <class Epi>
__global__ __launch_bounds__(256) void gemm_nt(
    const __bf16* __restrict__ A, const __bf16* __restrict__ B,
    int M, int N, int K, int lda, int ldb,
    i64 aBatch, int aDiv, i64 bBatch, int bMod, Epi epi)
{
    int z = blockIdx.z;
    A += (i64)(z / aDiv) * aBatch;
    B += (i64)(z % bMod) * bBatch;

    int lane   = threadIdx.x & 31;
    int w      = threadIdx.x >> 5;
    int laneHi = lane >> 4;
    int ln     = lane & 15;

    int mBlk = blockIdx.y * 128 + (w & 1) * 64;   // 2 waves along M
    int nBlk = blockIdx.x * 128 + (w >> 1) * 32;  // 4 waves along N

    v8f acc[4][2] = {};

    for (int k0 = 0; k0 < K; k0 += 32) {
        v16bf a[4], b[2];
#pragma unroll
        for (int i = 0; i < 4; ++i)
            a[i] = load_a_frag(A + (i64)(mBlk + i * 16 + ln) * lda + k0, laneHi);
#pragma unroll
        for (int j = 0; j < 2; ++j)
            b[j] = load_b_frag(B + (i64)(nBlk + j * 16 + ln) * ldb + k0, laneHi);

        if (k0 + 64 < K) {  // uniform condition (SGPR), EXEC untouched
            int kp = k0 + 64;
#pragma unroll
            for (int i = 0; i < 4; ++i)
                prefetch_wgp(A + (i64)(mBlk + i * 16 + ln) * lda + kp + laneHi * 8);
#pragma unroll
            for (int j = 0; j < 2; ++j)
                prefetch_wgp(B + (i64)(nBlk + j * 16 + ln) * ldb + kp + laneHi * 16);
        }

#pragma unroll
        for (int i = 0; i < 4; ++i)
#pragma unroll
            for (int j = 0; j < 2; ++j)
                acc[i][j] = __builtin_amdgcn_wmma_f32_16x16x32_bf16(
                    false, a[i], false, b[j], (short)0, acc[i][j], false, false);
    }

#pragma unroll
    for (int i = 0; i < 4; ++i)
#pragma unroll
        for (int j = 0; j < 2; ++j) {
            int gm0 = mBlk + i * 16 + laneHi * 8;
            int gn  = nBlk + j * 16 + ln;
#pragma unroll
            for (int r = 0; r < 8; ++r)
                epi.store(z, gm0 + r, gn, acc[i][j][r]);
        }
}

// ---------------------------------------------------------------------------
// LayerNorm over D=512: one block (256 thr) per row; output bf16
// ---------------------------------------------------------------------------
__global__ __launch_bounds__(256) void ln_kernel(
    const float* __restrict__ X, const float* __restrict__ g,
    const float* __restrict__ bb, __bf16* __restrict__ Y)
{
    i64 row = blockIdx.x;
    const float* x = X + row * D_;
    int t = threadIdx.x;
    float v0 = x[t], v1 = x[t + 256];
    __shared__ float red[256];
    red[t] = v0 + v1;
    __syncthreads();
    for (int s = 128; s > 0; s >>= 1) { if (t < s) red[t] += red[t + s]; __syncthreads(); }
    float mu = red[0] * (1.0f / D_);
    __syncthreads();
    float d0 = v0 - mu, d1 = v1 - mu;
    red[t] = d0 * d0 + d1 * d1;
    __syncthreads();
    for (int s = 128; s > 0; s >>= 1) { if (t < s) red[t] += red[t + s]; __syncthreads(); }
    float rstd = rsqrtf(red[0] * (1.0f / D_) + 1e-6f);
    Y[row * D_ + t]        = (__bf16)(d0 * rstd * g[t] + bb[t]);
    Y[row * D_ + t + 256]  = (__bf16)(d1 * rstd * g[t + 256] + bb[t + 256]);
}

// ---------------------------------------------------------------------------
// Row softmax over 1024 cols, in place on f32 P; optional bf16 copy; optional
// causal mask (adds -1e9 where col > l, matching reference mask*(-INF)).
// One block (256 thr, 4 cols each) per row; row index % L gives l.
// ---------------------------------------------------------------------------
__global__ __launch_bounds__(256) void softmax_kernel(
    float* __restrict__ P, __bf16* __restrict__ Wbf, int causal)
{
    i64 row = blockIdx.x;
    int l = (int)(row & (L_ - 1));
    float* p = P + row * (i64)L_;
    int t = threadIdx.x;
    float4 x = *(float4*)(p + t * 4);
    if (causal) {
        int c = t * 4;
        if (c + 0 > l) x.x -= 1e9f;
        if (c + 1 > l) x.y -= 1e9f;
        if (c + 2 > l) x.z -= 1e9f;
        if (c + 3 > l) x.w -= 1e9f;
    }
    __shared__ float red[256];
    red[t] = fmaxf(fmaxf(x.x, x.y), fmaxf(x.z, x.w));
    __syncthreads();
    for (int s = 128; s > 0; s >>= 1) { if (t < s) red[t] = fmaxf(red[t], red[t + s]); __syncthreads(); }
    float m = red[0];
    __syncthreads();
    x.x = expf(x.x - m); x.y = expf(x.y - m); x.z = expf(x.z - m); x.w = expf(x.w - m);
    red[t] = x.x + x.y + x.z + x.w;
    __syncthreads();
    for (int s = 128; s > 0; s >>= 1) { if (t < s) red[t] += red[t + s]; __syncthreads(); }
    float inv = 1.0f / red[0];
    x.x *= inv; x.y *= inv; x.z *= inv; x.w *= inv;
    *(float4*)(p + t * 4) = x;
    if (Wbf) {
        __bf16* q = Wbf + row * (i64)L_ + t * 4;
        q[0] = (__bf16)x.x; q[1] = (__bf16)x.y; q[2] = (__bf16)x.z; q[3] = (__bf16)x.w;
    }
}

// ---------------------------------------------------------------------------
// f32 -> bf16 convert (grid-stride)
// ---------------------------------------------------------------------------
__global__ __launch_bounds__(256) void cvt_kernel(
    const float* __restrict__ X, __bf16* __restrict__ Y, i64 n)
{
    for (i64 i = (i64)blockIdx.x * 256 + threadIdx.x; i < n; i += (i64)gridDim.x * 256)
        Y[i] = (__bf16)X[i];
}

// ---------------------------------------------------------------------------
// Host orchestration
// ---------------------------------------------------------------------------
extern "C" void kernel_launch(void* const* d_in, const int* in_sizes, int n_in,
                              void* d_out, int out_size, void* d_ws, size_t ws_size,
                              hipStream_t stream)
{
    const float* tokens = (const float*)d_in[0];
    // d_in[1] inp_lens, d_in[2] self_mask: unused (mask is causal, recomputed)
    const float* ln1_g = (const float*)d_in[3];
    const float* ln1_b = (const float*)d_in[4];
    const float* ln2_g = (const float*)d_in[5];
    const float* ln2_b = (const float*)d_in[6];
    const float* wq = (const float*)d_in[7];
    const float* wk = (const float*)d_in[8];
    const float* wv = (const float*)d_in[9];
    const float* wo = (const float*)d_in[10];
    const float* w1 = (const float*)d_in[11];
    const float* b1 = (const float*)d_in[12];
    const float* w2 = (const float*)d_in[13];
    const float* b2 = (const float*)d_in[14];
    const float* cwk = (const float*)d_in[15];
    const float* cwq = (const float*)d_in[17];

    float* out    = (float*)d_out;
    float* tOut   = out;                               // [B,L,D]
    float* selfW  = out + (i64)B_ * L_ * D_;           // [B,H,L,L]
    float* crossW = selfW + (i64)B_ * H_ * L_ * L_;    // [B,H,L,L]

    const size_t MB = 1ull << 20;
    char* w8 = (char*)d_ws;
    auto bf = [&](size_t off) { return (__bf16*)(w8 + off); };
    // live-range-aware bump layout (peak 168 MB)
    __bf16* xln   = bf(0 * MB);     //  4 MB  LN1(tokens)
    __bf16* tokbf = bf(4 * MB);     //  4 MB  bf16 tokens (cross-K A operand)
    __bf16* qbf   = bf(8 * MB);     // 32 MB
    __bf16* kbf   = bf(40 * MB);    // 32 MB
    __bf16* vT    = bf(72 * MB);    // 32 MB  V transposed [b,h][D][L]
    __bf16* Wbf   = bf(104 * MB);   // 64 MB  self_W in bf16
    __bf16* comb  = bf(8 * MB);     // 32 MB  (aliases dead q)
    __bf16* ybf   = bf(40 * MB);    //  4 MB  (aliases dead k)
    __bf16* h1bf  = bf(44 * MB);    // 16 MB
    __bf16* tbf   = bf(60 * MB);    //  4 MB  post-FFN t in bf16
    __bf16* ckbf  = bf(64 * MB);    // 32 MB  (aliases dead vT)
    __bf16* cqbf  = bf(96 * MB);    // 32 MB  (aliases dead Wbf)
    __bf16* wqb   = bf(128 * MB);   //  4 MB weights in bf16
    __bf16* wkb   = bf(132 * MB);
    __bf16* wvb   = bf(136 * MB);
    __bf16* wob   = bf(140 * MB);
    __bf16* w1b   = bf(144 * MB);
    __bf16* w2b   = bf(146 * MB);
    __bf16* cwkb  = bf(148 * MB);
    __bf16* cwqb  = bf(152 * MB);

    const float scale = 0.044194173824159216f;  // 1/sqrt(512)
    const i64 LD = (i64)L_ * D_, LL = (i64)L_ * L_, DD = (i64)D_ * D_;
    const i64 LF = (i64)L_ * F_, LHD = (i64)L_ * H_ * D_;

    auto cvt = [&](const float* s, __bf16* d, i64 n) {
        cvt_kernel<<<dim3(2048), dim3(256), 0, stream>>>(s, d, n);
    };
    cvt(tokens, tokbf, (i64)B_ * LD);
    cvt(wq, wqb, (i64)H_ * DD);  cvt(wk, wkb, (i64)H_ * DD);
    cvt(wv, wvb, (i64)H_ * DD);  cvt(wo, wob, (i64)D_ * H_ * D_);
    cvt(w1, w1b, (i64)F_ * D_);  cvt(w2, w2b, (i64)D_ * F_);
    cvt(cwk, cwkb, (i64)H_ * DD); cvt(cwq, cwqb, (i64)H_ * DD);

    // --- LN1 ---
    ln_kernel<<<dim3(B_ * L_), dim3(256), 0, stream>>>(tokens, ln1_g, ln1_b, xln);

    // --- Q, K (per b,h: z = b*8+h), V stored transposed ---
    dim3 gProj(D_ / 128, L_ / 128, B_ * H_);
    gemm_nt<EpiBf16><<<gProj, 256, 0, stream>>>(xln, wqb, L_, D_, D_, D_, D_,
        LD, H_, DD, H_, EpiBf16{qbf, LD, D_});
    gemm_nt<EpiBf16><<<gProj, 256, 0, stream>>>(xln, wkb, L_, D_, D_, D_, D_,
        LD, H_, DD, H_, EpiBf16{kbf, LD, D_});
    gemm_nt<EpiBf16T><<<gProj, 256, 0, stream>>>(xln, wvb, L_, D_, D_, D_, D_,
        LD, H_, DD, H_, EpiBf16T{vT, LD, L_});

    // --- self logits = Q K^T * scale -> d_out (f32) ---
    dim3 gAtt(L_ / 128, L_ / 128, B_ * H_);
    gemm_nt<EpiScaleF32><<<gAtt, 256, 0, stream>>>(qbf, kbf, L_, L_, D_, D_, D_,
        LD, 1, LD, B_ * H_, EpiScaleF32{selfW, LL, L_, scale});

    // --- causal softmax in place + bf16 copy ---
    softmax_kernel<<<dim3(B_ * H_ * L_), dim3(256), 0, stream>>>(selfW, Wbf, 1);

    // --- ctx = W V, written directly in concat/view-permuted comb layout ---
    gemm_nt<EpiComb><<<dim3(D_ / 128, L_ / 128, B_ * H_), 256, 0, stream>>>(
        Wbf, vT, L_, D_, L_, L_, L_, LL, 1, (i64)D_ * L_, B_ * H_, EpiComb{comb});

    // --- t = tokens + comb @ wo^T ---
    gemm_nt<EpiResF32><<<dim3(D_ / 128, L_ / 128, B_), 256, 0, stream>>>(
        comb, wob, L_, D_, H_ * D_, H_ * D_, H_ * D_,
        LHD, 1, 0, 1, EpiResF32{tOut, tokens, LD, D_});

    // --- FFN ---
    ln_kernel<<<dim3(B_ * L_), dim3(256), 0, stream>>>(tOut, ln2_g, ln2_b, ybf);
    gemm_nt<EpiBiasGelu><<<dim3(F_ / 128, L_ / 128, B_), 256, 0, stream>>>(
        ybf, w1b, L_, F_, D_, D_, D_, LD, 1, 0, 1, EpiBiasGelu{h1bf, b1, LF, F_});
    gemm_nt<EpiFFNOut><<<dim3(D_ / 128, L_ / 128, B_), 256, 0, stream>>>(
        h1bf, w2b, L_, D_, F_, F_, F_, LF, 1, 0, 1, EpiFFNOut{tOut, tbf, b2, LD, D_});

    // --- cross attention maps: K from tokens, Q from post-FFN t ---
    gemm_nt<EpiBf16><<<gProj, 256, 0, stream>>>(tokbf, cwkb, L_, D_, D_, D_, D_,
        LD, H_, DD, H_, EpiBf16{ckbf, LD, D_});
    gemm_nt<EpiBf16><<<gProj, 256, 0, stream>>>(tbf, cwqb, L_, D_, D_, D_, D_,
        LD, H_, DD, H_, EpiBf16{cqbf, LD, D_});
    gemm_nt<EpiScaleF32><<<gAtt, 256, 0, stream>>>(cqbf, ckbf, L_, L_, D_, D_, D_,
        LD, 1, LD, B_ * H_, EpiScaleF32{crossW, LL, L_, scale});
    softmax_kernel<<<dim3(B_ * H_ * L_), dim3(256), 0, stream>>>(crossW, nullptr, 0);

    (void)in_sizes; (void)n_in; (void)out_size; (void)ws_size;
}
